// TriangularUpdate_55173149884462
// MI455X (gfx1250) — compile-verified, matrix operand order checked by reading, not run on hardware
//
#include <hip/hip_runtime.h>
#include <hip/hip_bf16.h>
#include <stdint.h>

#define NN 512
#define DD 128

typedef unsigned short u16;
typedef unsigned int   u32;
typedef __attribute__((ext_vector_type(16))) __bf16 v16bf;
typedef __attribute__((ext_vector_type(8)))  float  v8f;

union AFrag { u32 u[8]; v16bf v; };

#if __has_builtin(__builtin_amdgcn_global_load_async_to_lds_b128)
#define HAVE_ASYNC_LDS 1
// Exact parameter types per hipcc diagnostic: non-const pointer to
// int __attribute__((vector_size(16))) in global (AS1) / LDS (AS3).
typedef __attribute__((__vector_size__(16))) int i32x4;
typedef __attribute__((address_space(1))) i32x4 gas_i32x4;
typedef __attribute__((address_space(3))) i32x4 las_i32x4;
#endif

__device__ __forceinline__ void wait_async0() {
#if __has_builtin(__builtin_amdgcn_s_wait_asynccnt)
    __builtin_amdgcn_s_wait_asynccnt(0);
#else
    asm volatile("s_wait_asynccnt 0x0" ::: "memory");
#endif
}

// round-to-nearest-even f32 -> bf16
__device__ __forceinline__ u16 f2bf(float f) {
    u32 u = __float_as_uint(f);
    u += 0x7fffu + ((u >> 16) & 1u);
    return (u16)(u >> 16);
}

// K index (even, covers pair k,k+1) held by VGPR-pair p for lane half `hi`
// per CDNA5 16-bit A/B fragment layout (ISA 7.12.2).
__device__ __forceinline__ int kpair(int p, int hi) {
    return (p < 4 ? 2 * p : 16 + 2 * (p - 4)) + (hi ? 8 : 0);
}

__device__ __forceinline__ float sigmoidf(float x) {
    return 1.0f / (1.0f + __expf(-x));
}

// ---------------------------------------------------------------------------
// Kernel W: convert weights f32 -> bf16 (row-major [n][k]) into workspace.
// ---------------------------------------------------------------------------
__global__ void wconv(const float* __restrict__ pi, const float* __restrict__ gi,
                      const float* __restrict__ po, const float* __restrict__ go,
                      u16* __restrict__ wpi, u16* __restrict__ wgi,
                      u16* __restrict__ wpo, u16* __restrict__ wgo) {
    int t = threadIdx.x;
    for (int i = t; i < DD * DD; i += 256) { wpi[i] = f2bf(pi[i]); wgi[i] = f2bf(gi[i]); }
    for (int i = t; i < DD * DD / 2; i += 256) { wpo[i] = f2bf(po[i]); wgo[i] = f2bf(go[i]); }
}

// ---------------------------------------------------------------------------
// Kernel A: LN(128) + gated projection (128->128) + mask, write bf16 planes.
// planes[n][r][c] for n<64 (a1,b1 row-major), planes[n][c][r] for n>=64
// (a2,b2 stored transposed so both einsums become C = P * Q^T).
// One workgroup = 16 tokens, 8 waves, each wave owns one 16-wide N-tile.
// ---------------------------------------------------------------------------
__global__ __launch_bounds__(256)
void fused_in(const float* __restrict__ x, const float* __restrict__ mask,
              const u16* __restrict__ wpi, const u16* __restrict__ wgi,
              const float* __restrict__ pib, const float* __restrict__ gib,
              const float* __restrict__ niw, const float* __restrict__ nib,
              u16* __restrict__ planes) {
    __shared__ __align__(16) float xs[16][DD];
    __shared__ float rs[16][16], rs2[16][16];
    __shared__ float mu_s[16], rsig_s[16];
    __shared__ float msk[16];
    __shared__ __align__(16) u16 hn[16][DD];

    const int tile = blockIdx.x;
    const long base = (long)tile * 16;
    const int tid = threadIdx.x;

    { // cooperative tile load: 16 tokens x 128 f32
        int tok = tid >> 4, seg = tid & 15;
        const float4* src = (const float4*)(x + (base + tok) * DD + seg * 8);
        float4 a = src[0], b = src[1];
        *(float4*)&xs[tok][seg * 8]     = a;
        *(float4*)&xs[tok][seg * 8 + 4] = b;
    }
    if (tid < 16) msk[tid] = mask[base + tid];
    __syncthreads();

    { // partial sums for mean/var: 16 threads per token, 8 elems each
        int tok = tid >> 4, part = tid & 15;
        float s = 0.f, s2 = 0.f;
        #pragma unroll
        for (int e = 0; e < 8; ++e) { float v = xs[tok][part * 8 + e]; s += v; s2 += v * v; }
        rs[tok][part] = s; rs2[tok][part] = s2;
    }
    __syncthreads();
    if (tid < 16) {
        float s = 0.f, s2 = 0.f;
        #pragma unroll
        for (int p = 0; p < 16; ++p) { s += rs[tid][p]; s2 += rs2[tid][p]; }
        float mu = s * (1.0f / DD);
        float var = s2 * (1.0f / DD) - mu * mu;
        mu_s[tid] = mu;
        rsig_s[tid] = rsqrtf(var + 1e-5f);
    }
    __syncthreads();
    { // normalize -> bf16 LDS tile
        int tok = tid >> 4, part = tid & 15;
        float mu = mu_s[tok], rr = rsig_s[tok];
        #pragma unroll
        for (int e = 0; e < 8; ++e) {
            int c = part * 8 + e;
            hn[tok][c] = f2bf((xs[tok][c] - mu) * rr * niw[c] + nib[c]);
        }
    }
    __syncthreads();

    const int wave = tid >> 5, lane = tid & 31;
    const int m = lane & 15, hi = lane >> 4;
    const int nb = wave * 16;

    v8f accp = {}, accg = {};
    #pragma unroll
    for (int kc = 0; kc < 4; ++kc) {
        AFrag a, bp, bg;
        #pragma unroll
        for (int p = 0; p < 8; ++p) {
            int k0 = kc * 32 + kpair(p, hi);
            a.u[p]  = *(const u32*)&hn[m][k0];
            bp.u[p] = *(const u32*)(wpi + (nb + m) * DD + k0);  // B[k][n] = W[n][k]
            bg.u[p] = *(const u32*)(wgi + (nb + m) * DD + k0);
        }
        accp = __builtin_amdgcn_wmma_f32_16x16x32_bf16(false, a.v, false, bp.v, (short)0, accp, false, false);
        accg = __builtin_amdgcn_wmma_f32_16x16x32_bf16(false, a.v, false, bg.v, (short)0, accg, false, false);
    }

    const int n = nb + m;
    const float pb = pib[n], gb = gib[n];
    const long pbase = (long)n * NN * NN;
    #pragma unroll
    for (int v = 0; v < 8; ++v) {
        int M = v + 8 * hi;
        long T = base + M;
        float val = (accp[v] + pb) * sigmoidf(accg[v] + gb) * msk[M];
        int r = (int)(T >> 9), c = (int)(T & 511);
        long off = (n < 64) ? (pbase + (long)r * NN + c)
                            : (pbase + (long)c * NN + r);
        planes[off] = f2bf(val);
    }
}

// ---------------------------------------------------------------------------
// Kernel B: 64 batched 512x512x512 bf16 GEMMs, C = P * Q^T.
// Workgroup: 128x128 tile of one channel plane; double-buffered LDS K-chunks
// of 32 staged via gfx1250 GLOBAL_LOAD_ASYNC_TO_LDS_B128 (ASYNCcnt) when the
// builtin exists, else via register staging. Each wave owns a 32x64 C block:
// 2 A-frags x 4 B-frags -> 8 WMMAs per K-chunk (12 LDS loads / 8 WMMAs).
// Output f32 plane-major out12[d][i][j].
// ---------------------------------------------------------------------------
__global__ __launch_bounds__(256)
void tri_gemm(const u16* __restrict__ planes, float* __restrict__ out12) {
    const int jt = blockIdx.x, it = blockIdx.y, d = blockIdx.z;
    const int Pp = (d < 32) ? d : d + 32;        // a1[d] or a2[d-32] (transposed)
    const u16* __restrict__ P = planes + (long)Pp * NN * NN;
    const u16* __restrict__ Q = planes + (long)(Pp + 32) * NN * NN;

    __shared__ __align__(16) u16 As[2][128][32];
    __shared__ __align__(16) u16 Bs[2][128][32];

    const int tid = threadIdx.x;
    const int lrow = tid >> 1, co = (tid & 1) * 16;
    const u16* Pr = P + (long)(it * 128 + lrow) * NN + co;
    const u16* Qr = Q + (long)(jt * 128 + lrow) * NN + co;

#if defined(HAVE_ASYNC_LDS)
    // stage K-chunk kc into buffer b: each thread moves 32B of A and 32B of B
    #define STAGE(kc, b)                                                                    \
        do {                                                                                \
            __builtin_amdgcn_global_load_async_to_lds_b128(                                 \
                (gas_i32x4*)(Pr + (kc) * 32),     (las_i32x4*)&As[b][lrow][co],     0, 0);  \
            __builtin_amdgcn_global_load_async_to_lds_b128(                                 \
                (gas_i32x4*)(Pr + (kc) * 32 + 8), (las_i32x4*)&As[b][lrow][co + 8], 0, 0);  \
            __builtin_amdgcn_global_load_async_to_lds_b128(                                 \
                (gas_i32x4*)(Qr + (kc) * 32),     (las_i32x4*)&Bs[b][lrow][co],     0, 0);  \
            __builtin_amdgcn_global_load_async_to_lds_b128(                                 \
                (gas_i32x4*)(Qr + (kc) * 32 + 8), (las_i32x4*)&Bs[b][lrow][co + 8], 0, 0);  \
        } while (0)

    STAGE(0, 0);
    wait_async0();
    __syncthreads();
#else
    uint4 pa, pb, qa, qb;
    pa = *(const uint4*)(Pr);     pb = *(const uint4*)(Pr + 8);
    qa = *(const uint4*)(Qr);     qb = *(const uint4*)(Qr + 8);
    *(uint4*)&As[0][lrow][co] = pa;  *(uint4*)&As[0][lrow][co + 8] = pb;
    *(uint4*)&Bs[0][lrow][co] = qa;  *(uint4*)&Bs[0][lrow][co + 8] = qb;
    __syncthreads();
#endif

    const int wave = tid >> 5, lane = tid & 31;
    const int m = lane & 15, hi = lane >> 4;
    const int I0 = (wave & 3) * 32;   // 4 waves cover 128 rows (32 each)
    const int J0 = (wave >> 2) * 64;  // 2 wave groups cover 128 cols (64 each)

    v8f acc[2][4] = {};
    int buf = 0;
    for (int kc = 0; kc < 16; ++kc) {
#if defined(HAVE_ASYNC_LDS)
        if (kc < 15) STAGE(kc + 1, buf ^ 1);  // overlaps with compute below
#else
        if (kc < 15) { // stage next K-chunk in registers while computing
            const u16* Pn = Pr + (kc + 1) * 32;
            const u16* Qn = Qr + (kc + 1) * 32;
            pa = *(const uint4*)(Pn); pb = *(const uint4*)(Pn + 8);
            qa = *(const uint4*)(Qn); qb = *(const uint4*)(Qn + 8);
            if (kc < 14) { // gfx1250 global_prefetch_b8 two chunks ahead
                __builtin_prefetch(Pr + (kc + 2) * 32, 0, 1);
                __builtin_prefetch(Qr + (kc + 2) * 32, 0, 1);
            }
        }
#endif
        AFrag a[2];
        #pragma unroll
        for (int r = 0; r < 2; ++r)
            #pragma unroll
            for (int p = 0; p < 8; ++p)
                a[r].u[p] = *(const u32*)&As[buf][I0 + r * 16 + m][kpair(p, hi)];
        #pragma unroll
        for (int t = 0; t < 4; ++t) {
            AFrag b;
            #pragma unroll
            for (int p = 0; p < 8; ++p)
                b.u[p] = *(const u32*)&Bs[buf][J0 + t * 16 + m][kpair(p, hi)];
            #pragma unroll
            for (int r = 0; r < 2; ++r)
                acc[r][t] = __builtin_amdgcn_wmma_f32_16x16x32_bf16(
                    false, a[r].v, false, b.v, (short)0, acc[r][t], false, false);
        }
        if (kc < 15) {
#if defined(HAVE_ASYNC_LDS)
            wait_async0();     // our stage (kc+1) writes have landed in LDS
            __syncthreads();   // all waves done reading buf & done staging
#else
            __syncthreads();
            *(uint4*)&As[buf ^ 1][lrow][co] = pa;  *(uint4*)&As[buf ^ 1][lrow][co + 8] = pb;
            *(uint4*)&Bs[buf ^ 1][lrow][co] = qa;  *(uint4*)&Bs[buf ^ 1][lrow][co + 8] = qb;
            __syncthreads();
#endif
            buf ^= 1;
        }
    }

    const long obase = (long)d * NN * NN;
    #pragma unroll
    for (int r = 0; r < 2; ++r) {
        #pragma unroll
        for (int t = 0; t < 4; ++t) {
            int J = jt * 128 + J0 + t * 16 + m;
            #pragma unroll
            for (int v = 0; v < 8; ++v) {
                int I = it * 128 + I0 + r * 16 + v + 8 * hi;
                out12[obase + (long)I * NN + J] = acc[r][t][v];
            }
        }
    }
}

// ---------------------------------------------------------------------------
// Kernel C: gather 64 channels/token, LN(64), gated projection (64->128),
// write f32 output. One workgroup = 16 consecutive tokens (same row i).
// ---------------------------------------------------------------------------
__global__ __launch_bounds__(256)
void fused_out(const float* __restrict__ out12,
               const u16* __restrict__ wpo, const u16* __restrict__ wgo,
               const float* __restrict__ pob, const float* __restrict__ gob,
               const float* __restrict__ now, const float* __restrict__ nob,
               float* __restrict__ outp) {
    __shared__ float g12[16][64];
    __shared__ float rs[16][16], rs2[16][16];
    __shared__ float mu_s[16], rsig_s[16];
    __shared__ __align__(16) u16 hn[16][64];

    const int tile = blockIdx.x;
    const long base = (long)tile * 16;
    const int i = (int)(base >> 9), j0 = (int)(base & 511);
    const int tid = threadIdx.x;

    { // gather: thread -> (channel d, 4 consecutive j)
        int d = tid >> 2, q = tid & 3;
        float4 v = *(const float4*)(out12 + (long)d * NN * NN + (long)i * NN + j0 + q * 4);
        g12[q * 4 + 0][d] = v.x; g12[q * 4 + 1][d] = v.y;
        g12[q * 4 + 2][d] = v.z; g12[q * 4 + 3][d] = v.w;
    }
    __syncthreads();
    {
        int tok = tid >> 4, part = tid & 15;
        float s = 0.f, s2 = 0.f;
        #pragma unroll
        for (int e = 0; e < 4; ++e) { float v = g12[tok][part * 4 + e]; s += v; s2 += v * v; }
        rs[tok][part] = s; rs2[tok][part] = s2;
    }
    __syncthreads();
    if (tid < 16) {
        float s = 0.f, s2 = 0.f;
        #pragma unroll
        for (int p = 0; p < 16; ++p) { s += rs[tid][p]; s2 += rs2[tid][p]; }
        float mu = s * (1.0f / 64.0f);
        float var = s2 * (1.0f / 64.0f) - mu * mu;
        mu_s[tid] = mu;
        rsig_s[tid] = rsqrtf(var + 1e-5f);
    }
    __syncthreads();
    {
        int tok = tid >> 4, part = tid & 15;
        float mu = mu_s[tok], rr = rsig_s[tok];
        #pragma unroll
        for (int e = 0; e < 4; ++e) {
            int c = part * 4 + e;
            hn[tok][c] = f2bf((g12[tok][c] - mu) * rr * now[c] + nob[c]);
        }
    }
    __syncthreads();

    const int wave = tid >> 5, lane = tid & 31;
    const int m = lane & 15, hi = lane >> 4;
    const int nb = wave * 16;

    v8f accp = {}, accg = {};
    #pragma unroll
    for (int kc = 0; kc < 2; ++kc) {
        AFrag a, bp, bg;
        #pragma unroll
        for (int p = 0; p < 8; ++p) {
            int k0 = kc * 32 + kpair(p, hi);
            a.u[p]  = *(const u32*)&hn[m][k0];
            bp.u[p] = *(const u32*)(wpo + (nb + m) * 64 + k0);
            bg.u[p] = *(const u32*)(wgo + (nb + m) * 64 + k0);
        }
        accp = __builtin_amdgcn_wmma_f32_16x16x32_bf16(false, a.v, false, bp.v, (short)0, accp, false, false);
        accg = __builtin_amdgcn_wmma_f32_16x16x32_bf16(false, a.v, false, bg.v, (short)0, accg, false, false);
    }

    const int n = nb + m;
    const float pb = pob[n], gb = gob[n];
    #pragma unroll
    for (int v = 0; v < 8; ++v) {
        int M = v + 8 * hi;
        long T = base + M;
        outp[T * DD + n] = (accp[v] + pb) * sigmoidf(accg[v] + gb);
    }
}

// ---------------------------------------------------------------------------
extern "C" void kernel_launch(void* const* d_in, const int* in_sizes, int n_in,
                              void* d_out, int out_size, void* d_ws, size_t ws_size,
                              hipStream_t stream) {
    (void)in_sizes; (void)n_in; (void)out_size; (void)ws_size;
    const float* x    = (const float*)d_in[0];
    const float* mask = (const float*)d_in[1];
    const float* pi_w = (const float*)d_in[2];
    const float* gi_w = (const float*)d_in[3];
    const float* pi_b = (const float*)d_in[4];
    const float* gi_b = (const float*)d_in[5];
    const float* po_w = (const float*)d_in[6];
    const float* go_w = (const float*)d_in[7];
    const float* po_b = (const float*)d_in[8];
    const float* go_b = (const float*)d_in[9];
    const float* ni_w = (const float*)d_in[10];
    const float* ni_b = (const float*)d_in[11];
    const float* no_w = (const float*)d_in[12];
    const float* no_b = (const float*)d_in[13];

    char* ws = (char*)d_ws;
    u16*   planes = (u16*)ws;                             // 128 planes bf16: 64 MiB
    float* out12  = (float*)(ws + 67108864LL);            // 64 f32 planes:  64 MiB
    u16*   wpi    = (u16*)(ws + 134217728LL);             // bf16 weights
    u16*   wgi    = wpi + DD * DD;
    u16*   wpo    = wgi + DD * DD;
    u16*   wgo    = wpo + DD * DD / 2;

    wconv<<<1, 256, 0, stream>>>(pi_w, gi_w, po_w, go_w, wpi, wgi, wpo, wgo);
    fused_in<<<16384, 256, 0, stream>>>(x, mask, wpi, wgi, pi_b, gi_b, ni_w, ni_b, planes);
    tri_gemm<<<dim3(4, 4, 64), 256, 0, stream>>>(planes, out12);
    fused_out<<<16384, 256, 0, stream>>>(out12, wpo, wgo, po_b, go_b, no_w, no_b, (float*)d_out);
}
